// FLOT_12850542149708
// MI455X (gfx1250) — matrix operand order repaired and checked by reading, not compiled, not run on hardware
//
#include <hip/hip_runtime.h>
#include <hip/hip_bf16.h>

// ---------------------------------------------------------------------------
// FLOT scene flow on gfx1250 (MI455X). All GEMMs via v_wmma_f32_16x16x32_f16.
// ---------------------------------------------------------------------------

typedef __attribute__((ext_vector_type(16))) _Float16 v16h;
typedef __attribute__((ext_vector_type(8)))  _Float16 h8;
typedef __attribute__((ext_vector_type(4)))  _Float16 h4;
typedef __attribute__((ext_vector_type(8)))  float    v8f;

#define NPTS 4096
#define BATCH 2
#define BN (BATCH * NPTS)
#define KNN 32
#define FDIM 128

__device__ __forceinline__ v8f wmma16(v16h a, v16h b, v8f c) {
  return __builtin_amdgcn_wmma_f32_16x16x32_f16(false, a, false, b, (short)0, c,
                                                false, false);
}

// A fragment (16x32 f16): lane<16 -> row=lane, K {0..7, 16..23};
//                         lane>=16 -> row=lane-16, K {8..15, 24..31}.
// base points at (row0, k0); ld in halves.
__device__ __forceinline__ v16h load_frag_A(const _Float16* base, int ld, int lane) {
  int r = lane & 15, h = lane >> 4;
  const _Float16* p = base + r * ld + h * 8;
  union { v16h v; h8 q[2]; } u;
  u.q[0] = *(const h8*)p;
  u.q[1] = *(const h8*)(p + 16);
  return u.v;
}

// B fragment (32x16 f16) stored as B[k][n] = mat[n][k] (row-major [n, ld]):
// lane<16 -> n=lane, K {0..7, 16..23}; lane>=16 -> n=lane-16, K {8..15, 24..31}.
__device__ __forceinline__ v16h load_frag_B(const _Float16* base, int ld, int lane) {
  int r = lane & 15, h = lane >> 4;
  const _Float16* p = base + r * ld + h * 8;
  union { v16h v; h8 q[2]; } u;
  u.q[0] = *(const h8*)p;
  u.q[1] = *(const h8*)(p + 16);
  return u.v;
}

__device__ __forceinline__ float lrelu(float x) { return x >= 0.f ? x : 0.1f * x; }

// ---------------------------------------------------------------------------
// Weight prep: W f32 [K, C] -> Wt f16 [C, Kpad], zero padded.
// ---------------------------------------------------------------------------
__global__ void transw_kernel(const float* __restrict__ W, _Float16* __restrict__ Wt,
                              int K, int C, int Kpad) {
  int t = blockIdx.x * 256 + threadIdx.x;
  if (t >= C * Kpad) return;
  int c = t / Kpad, k = t % Kpad;
  Wt[t] = (k < K) ? (_Float16)W[k * C + c] : (_Float16)0.f;
}

// ---------------------------------------------------------------------------
// KNN: for each query point, 32 nearest (incl. self) + edge features.
// ---------------------------------------------------------------------------
__global__ __launch_bounds__(128) void knn_kernel(const float* __restrict__ pc,
                                                  int* __restrict__ idxo,
                                                  float* __restrict__ efo) {
  __shared__ float tx[128], ty[128], tz[128];
  int q = blockIdx.x * 128 + threadIdx.x;   // global point id
  int b = q / NPTS, i = q % NPTS;
  const float* pcb = pc + (size_t)b * NPTS * 3;
  float qx = pcb[i * 3 + 0], qy = pcb[i * 3 + 1], qz = pcb[i * 3 + 2];
  float bd[KNN];
  int   bi[KNN];
#pragma unroll
  for (int t = 0; t < KNN; ++t) { bd[t] = 3.4e38f; bi[t] = 0; }
  for (int tile = 0; tile < NPTS; tile += 128) {
    __syncthreads();
    int j = tile + threadIdx.x;
    tx[threadIdx.x] = pcb[j * 3 + 0];
    ty[threadIdx.x] = pcb[j * 3 + 1];
    tz[threadIdx.x] = pcb[j * 3 + 2];
    __syncthreads();
    for (int u = 0; u < 128; ++u) {
      float dx = tx[u] - qx, dy = ty[u] - qy, dz = tz[u] - qz;
      float d = dx * dx + dy * dy + dz * dz;
      if (d < bd[KNN - 1]) {
        bd[KNN - 1] = d; bi[KNN - 1] = tile + u;
#pragma unroll
        for (int t = KNN - 1; t > 0; --t) {
          if (bd[t] < bd[t - 1]) {
            float td = bd[t]; bd[t] = bd[t - 1]; bd[t - 1] = td;
            int   ti = bi[t]; bi[t] = bi[t - 1]; bi[t - 1] = ti;
          }
        }
      }
    }
  }
#pragma unroll
  for (int t = 0; t < KNN; ++t) {
    int j = bi[t];
    idxo[(size_t)q * KNN + t] = j;
    efo[((size_t)q * KNN + t) * 3 + 0] = pcb[j * 3 + 0] - qx;
    efo[((size_t)q * KNN + t) * 3 + 1] = pcb[j * 3 + 1] - qy;
    efo[((size_t)q * KNN + t) * 3 + 2] = pcb[j * 3 + 2] - qz;
  }
}

// ---------------------------------------------------------------------------
// SetConv fc1 + max-pool (fused). One wave per point: gather 32 neighbor rows
// (concat x[j] | ef) into LDS as f16, WMMA against transposed weights, lrelu,
// max over 32 neighbors.
// ---------------------------------------------------------------------------
__global__ __launch_bounds__(256) void fc1max_kernel(
    const float* __restrict__ x, int Cin, const int* __restrict__ idx,
    const float* __restrict__ ef, const _Float16* __restrict__ Wt, int Kpad,
    const float* __restrict__ bias, int Cout, float* __restrict__ out) {
  __shared__ _Float16 sA[8 * 32 * 96];
  int wid = threadIdx.x >> 5, lane = threadIdx.x & 31;
  int p = blockIdx.x * 8 + wid;          // global point in [0, BN)
  int b = p >> 12;
  _Float16* rowbuf = sA + wid * 32 * 96;

  // stage: one lane per neighbor row
  {
    int j = idx[(size_t)p * KNN + lane];
    const float* xr = x + ((size_t)b * NPTS + j) * Cin;
    const float* e = ef + ((size_t)p * KNN + lane) * 3;
    _Float16* r = rowbuf + lane * Kpad;
    int c = 0;
    for (; c < Cin; ++c) r[c] = (_Float16)xr[c];
    r[Cin + 0] = (_Float16)e[0];
    r[Cin + 1] = (_Float16)e[1];
    r[Cin + 2] = (_Float16)e[2];
    for (c = Cin + 3; c < Kpad; ++c) r[c] = (_Float16)0.f;
  }
  __syncthreads();

  for (int ct = 0; ct < Cout; ct += 16) {
    v8f acc0 = {}, acc1 = {};
    for (int kb = 0; kb < Kpad; kb += 32) {
      v16h Bf = load_frag_B(Wt + (size_t)ct * Kpad + kb, Kpad, lane);
      v16h A0 = load_frag_A(rowbuf + kb, Kpad, lane);
      v16h A1 = load_frag_A(rowbuf + 16 * Kpad + kb, Kpad, lane);
      acc0 = wmma16(A0, Bf, acc0);
      acc1 = wmma16(A1, Bf, acc1);
    }
    union { v8f v; float f[8]; } u0, u1;
    u0.v = acc0; u1.v = acc1;
    float m = -3.4e38f;
#pragma unroll
    for (int j = 0; j < 8; ++j) m = fmaxf(m, fmaxf(u0.f[j], u1.f[j]));
    m = fmaxf(m, __shfl_xor(m, 16, 32));   // combine the two M halves
    if (lane < 16) {
      int n = ct + lane;
      out[(size_t)p * Cout + n] = lrelu(m + bias[n]);
    }
  }
}

// ---------------------------------------------------------------------------
// Dense GEMM + bias + lrelu: X f32 [8192, K] x Wt f16 [C, K] -> out f32.
// One wave per 16-row tile, looping column tiles. K in {32, 64, 128}.
// ---------------------------------------------------------------------------
__global__ __launch_bounds__(256) void gemm_lrelu_kernel(
    const float* __restrict__ X, int K, const _Float16* __restrict__ Wt,
    const float* __restrict__ bias, int C, float* __restrict__ out) {
  __shared__ _Float16 sA[8 * 16 * 128];
  int wid = threadIdx.x >> 5, lane = threadIdx.x & 31;
  int row0 = (blockIdx.x * 8 + wid) * 16;
  _Float16* rb = sA + wid * 16 * 128;

  {  // stage 16 rows f32->f16 (half a row per lane, vectorized)
    int rr = lane & 15, hh = lane >> 4;
    const float4* xr4 = (const float4*)(X + (size_t)(row0 + rr) * K + hh * (K / 2));
    h4* d4 = (h4*)(rb + rr * K + hh * (K / 2));
    for (int c = 0; c < K / 8; ++c) {
      float4 v = xr4[c];
      h4 o = {(_Float16)v.x, (_Float16)v.y, (_Float16)v.z, (_Float16)v.w};
      d4[c] = o;
    }
  }
  __syncthreads();

  int h = lane >> 4;
  for (int ct = 0; ct < C; ct += 16) {
    v8f acc = {};
    for (int kb = 0; kb < K; kb += 32)
      acc = wmma16(load_frag_A(rb + kb, K, lane),
                   load_frag_B(Wt + (size_t)ct * K + kb, K, lane), acc);
    union { v8f v; float f[8]; } u; u.v = acc;
    int n = ct + (lane & 15);
    float bn = bias[n];
#pragma unroll
    for (int j = 0; j < 8; ++j)
      out[(size_t)(row0 + j + 8 * h) * C + n] = lrelu(u.f[j] + bn);
  }
}

// ---------------------------------------------------------------------------
// L2 normalize f32 [BN,128] -> f16 (wave per point)
// ---------------------------------------------------------------------------
__global__ __launch_bounds__(128) void normalize_kernel(const float* __restrict__ f,
                                                        _Float16* __restrict__ fh) {
  int wid = threadIdx.x >> 5, lane = threadIdx.x & 31;
  int p = blockIdx.x * 4 + wid;
  const float4 v = *(const float4*)(f + (size_t)p * FDIM + lane * 4);
  float s = v.x * v.x + v.y * v.y + v.z * v.z + v.w * v.w;
#pragma unroll
  for (int m = 16; m >= 1; m >>= 1) s += __shfl_xor(s, m, 32);
  float inv = rsqrtf(s + 1e-8f);
  h4 o = {(_Float16)(v.x * inv), (_Float16)(v.y * inv),
          (_Float16)(v.z * inv), (_Float16)(v.w * inv)};
  *(h4*)(fh + (size_t)p * FDIM + lane * 4) = o;
}

// ---------------------------------------------------------------------------
// K matrix: K[b,n,m] = exp(-(1 - f1.f2)/eps) * (dist^2 < 100).
// One wave per 16x128 strip: A fragments (16 rows x 128 of f1) stay in
// registers; sweep 8 B tiles -> 32 WMMAs per A load.
// ---------------------------------------------------------------------------
__global__ __launch_bounds__(256) void cost_kernel(
    const _Float16* __restrict__ f1h, const _Float16* __restrict__ f2h,
    const float* __restrict__ pc1, const float* __restrict__ pc2,
    const float* __restrict__ p_eps, float* __restrict__ Kmat) {
  int wid = threadIdx.x >> 5, lane = threadIdx.x & 31;
  int t = blockIdx.x * 8 + wid;              // b*8192 + tn*32 + mb
  int b = t >> 13, rem = t & 8191;
  int n0 = (rem >> 5) * 16, m0 = (rem & 31) * 128;

  const _Float16* Ab = f1h + ((size_t)b * NPTS + n0) * FDIM;
  v16h Af[4];
#pragma unroll
  for (int s = 0; s < 4; ++s) Af[s] = load_frag_A(Ab + s * 32, FDIM, lane);

  float eps = expf(p_eps[0]) + 0.03f;
  int h = lane >> 4;
  float p1x[8], p1y[8], p1z[8];
#pragma unroll
  for (int j = 0; j < 8; ++j) {
    int n = n0 + j + 8 * h;
    const float* p1 = pc1 + ((size_t)b * NPTS + n) * 3;
    p1x[j] = p1[0]; p1y[j] = p1[1]; p1z[j] = p1[2];
  }

  for (int mt = 0; mt < 8; ++mt) {
    int mbase = m0 + mt * 16;
    const _Float16* Bb = f2h + ((size_t)b * NPTS + mbase) * FDIM;
    v8f acc = {};
#pragma unroll
    for (int s = 0; s < 4; ++s)
      acc = wmma16(Af[s], load_frag_B(Bb + s * 32, FDIM, lane), acc);

    int m = mbase + (lane & 15);
    const float* p2 = pc2 + ((size_t)b * NPTS + m) * 3;
    float p2x = p2[0], p2y = p2[1], p2z = p2[2];
    union { v8f v; float f[8]; } u; u.v = acc;
#pragma unroll
    for (int j = 0; j < 8; ++j) {
      int n = n0 + j + 8 * h;
      float dx = p1x[j] - p2x, dy = p1y[j] - p2y, dz = p1z[j] - p2z;
      float dsq = dx * dx + dy * dy + dz * dz;
      float c = 1.f - u.f[j];
      float kv = (dsq < 100.f) ? expf(-c / eps) : 0.f;
      Kmat[((size_t)b * NPTS + n) * NPTS + m] = kv;
    }
  }
}

// ---------------------------------------------------------------------------
// Sinkhorn helpers
// ---------------------------------------------------------------------------
__global__ void set_const_kernel(float* __restrict__ v, float c, int n) {
  int t = blockIdx.x * 256 + threadIdx.x;
  if (t < n) v[t] = c;
}

// partial column sums: KTa[m] += sum_n K[n,m]*a[n] (atomic, float4 columns).
// grid = B * 32 nchunks * 4 mchunks; block = 256 threads x 4 columns each.
__global__ __launch_bounds__(256) void colsum_kernel(const float* __restrict__ Kmat,
                                                     const float* __restrict__ avec,
                                                     float* __restrict__ kta) {
  int t = blockIdx.x;
  int b = t / 128, r = t % 128, nc = r / 4, mc = r % 4;
  int m4 = mc * 256 + threadIdx.x;                 // float4 column index
  const float4* Kb4 = (const float4*)(Kmat + (size_t)b * NPTS * NPTS);
  const float* ab = avec + b * NPTS;
  float sx = 0.f, sy = 0.f, sz = 0.f, sw = 0.f;
  int n0 = nc * 128;
  for (int n = n0; n < n0 + 128; ++n) {
    __builtin_prefetch(&Kb4[(size_t)(n + 2) * 1024 + m4], 0, 0);
    float av = ab[n];
    float4 kv = Kb4[(size_t)n * 1024 + m4];
    sx += kv.x * av; sy += kv.y * av; sz += kv.z * av; sw += kv.w * av;
  }
  float* o = kta + b * NPTS + m4 * 4;
  atomicAdd(o + 0, sx); atomicAdd(o + 1, sy);
  atomicAdd(o + 2, sz); atomicAdd(o + 3, sw);
}

__global__ void bupdate_kernel(const float* __restrict__ kta, float* __restrict__ bvec,
                               const float* __restrict__ p_eps,
                               const float* __restrict__ p_gam) {
  int t = blockIdx.x * 256 + threadIdx.x;   // [0, BN)
  float eps = expf(p_eps[0]) + 0.03f;
  float gam = expf(p_gam[0]);
  float power = gam / (gam + eps);
  float v = (1.0f / NPTS) / (kta[t] + 1e-8f);
  bvec[t] = powf(v, power);
}

// Kb[n] = sum_m K[n,m]*b[m]; a[n] = (prob1/(Kb+1e-8))^power. Block per row.
__global__ __launch_bounds__(256) void rowa_kernel(const float* __restrict__ Kmat,
                                                   const float* __restrict__ bvec,
                                                   float* __restrict__ avec,
                                                   const float* __restrict__ p_eps,
                                                   const float* __restrict__ p_gam) {
  __shared__ float red[256];
  int row = blockIdx.x;                     // b*NPTS + n
  int b = row >> 12;
  const float4* Kr4 = (const float4*)(Kmat + (size_t)row * NPTS);
  const float4* bb4 = (const float4*)(bvec + b * NPTS);
  float s = 0.f;
  for (int m4 = threadIdx.x; m4 < 1024; m4 += 256) {
    __builtin_prefetch(&Kr4[m4 + 256], 0, 0);
    float4 kv = Kr4[m4], bv = bb4[m4];
    s += kv.x * bv.x + kv.y * bv.y + kv.z * bv.z + kv.w * bv.w;
  }
  red[threadIdx.x] = s; __syncthreads();
  for (int st = 128; st > 0; st >>= 1) {
    if (threadIdx.x < st) red[threadIdx.x] += red[threadIdx.x + st];
    __syncthreads();
  }
  if (threadIdx.x == 0) {
    float eps = expf(p_eps[0]) + 0.03f;
    float gam = expf(p_gam[0]);
    float power = gam / (gam + eps);
    avec[row] = powf((1.0f / NPTS) / (red[0] + 1e-8f), power);
  }
}

// ot_flow[n] = (sum_m a*K*b*pc2[m]) / (sum_m a*K*b + 1e-8) - pc1[n]
__global__ __launch_bounds__(256) void flow_kernel(const float* __restrict__ Kmat,
                                                   const float* __restrict__ avec,
                                                   const float* __restrict__ bvec,
                                                   const float* __restrict__ pc1,
                                                   const float* __restrict__ pc2,
                                                   float* __restrict__ otflow) {
  __shared__ float red[4][256];
  int row = blockIdx.x;
  int b = row >> 12;
  const float4* Kr4 = (const float4*)(Kmat + (size_t)row * NPTS);
  const float4* bb4 = (const float4*)(bvec + b * NPTS);
  const float4* p24 = (const float4*)(pc2 + (size_t)b * NPTS * 3);
  float s = 0.f, px = 0.f, py = 0.f, pz = 0.f;
  for (int m4 = threadIdx.x; m4 < 1024; m4 += 256) {
    __builtin_prefetch(&Kr4[m4 + 256], 0, 0);
    float4 kv = Kr4[m4], bv = bb4[m4];
    float k0 = kv.x * bv.x, k1 = kv.y * bv.y, k2 = kv.z * bv.z, k3 = kv.w * bv.w;
    float4 q0 = p24[m4 * 3 + 0];   // x0 y0 z0 x1
    float4 q1 = p24[m4 * 3 + 1];   // y1 z1 x2 y2
    float4 q2 = p24[m4 * 3 + 2];   // z2 x3 y3 z3
    s += k0 + k1 + k2 + k3;
    px += k0 * q0.x + k1 * q0.w + k2 * q1.z + k3 * q2.y;
    py += k0 * q0.y + k1 * q1.x + k2 * q1.w + k3 * q2.z;
    pz += k0 * q0.z + k1 * q1.y + k2 * q2.x + k3 * q2.w;
  }
  red[0][threadIdx.x] = s;  red[1][threadIdx.x] = px;
  red[2][threadIdx.x] = py; red[3][threadIdx.x] = pz;
  __syncthreads();
  for (int st = 128; st > 0; st >>= 1) {
    if (threadIdx.x < st)
#pragma unroll
      for (int c = 0; c < 4; ++c) red[c][threadIdx.x] += red[c][threadIdx.x + st];
    __syncthreads();
  }
  if (threadIdx.x == 0) {
    float a = avec[row];
    float inv = 1.f / (a * red[0][0] + 1e-8f);
    otflow[(size_t)row * 3 + 0] = a * red[1][0] * inv - pc1[(size_t)row * 3 + 0];
    otflow[(size_t)row * 3 + 1] = a * red[2][0] * inv - pc1[(size_t)row * 3 + 1];
    otflow[(size_t)row * 3 + 2] = a * red[3][0] * inv - pc1[(size_t)row * 3 + 2];
  }
}

// out = ot_flow + h3 @ Wfc[128,3] + bfc
__global__ void finalfc_kernel(const float* __restrict__ h3, const float* __restrict__ W,
                               const float* __restrict__ bfc,
                               const float* __restrict__ otflow, float* __restrict__ out) {
  int p = blockIdx.x * 256 + threadIdx.x;   // [0, BN)
  const float* x = h3 + (size_t)p * FDIM;
  float a0 = bfc[0], a1 = bfc[1], a2 = bfc[2];
  for (int k = 0; k < FDIM; ++k) {
    float xv = x[k];
    a0 += xv * W[k * 3 + 0];
    a1 += xv * W[k * 3 + 1];
    a2 += xv * W[k * 3 + 2];
  }
  out[(size_t)p * 3 + 0] = otflow[(size_t)p * 3 + 0] + a0;
  out[(size_t)p * 3 + 1] = otflow[(size_t)p * 3 + 1] + a1;
  out[(size_t)p * 3 + 2] = otflow[(size_t)p * 3 + 2] + a2;
}

// ---------------------------------------------------------------------------
// Host orchestration
// ---------------------------------------------------------------------------
extern "C" void kernel_launch(void* const* d_in, const int* in_sizes, int n_in,
                              void* d_out, int out_size, void* d_ws, size_t ws_size,
                              hipStream_t stream) {
  // d_in order (setup_inputs dict insertion order, nested dicts flattened):
  // 0 pc1, 1 pc2, 2 gamma, 3 epsilon,
  // 4.. feat_conv1{fc1.W,fc1.b,fc2.W,fc2.b,fc3.W,fc3.b}, 10.. feat_conv2,
  // 16.. feat_conv3, 22.. ref_conv1, 28.. ref_conv2, 34.. ref_conv3, 40 fc.W, 41 fc.b
  const float* pc1 = (const float*)d_in[0];
  const float* pc2 = (const float*)d_in[1];
  const float* p_gam = (const float*)d_in[2];
  const float* p_eps = (const float*)d_in[3];

  auto alignup = [](size_t x) { return (x + 255) & ~(size_t)255; };
  char* ws = (char*)d_ws;
  size_t off = 0;
  auto carve = [&](size_t bytes) { char* p = ws + off; off += alignup(bytes); return p; };

  int*   idx1 = (int*)  carve((size_t)BN * KNN * 4);
  float* ef1  = (float*)carve((size_t)BN * KNN * 3 * 4);
  int*   idx2 = (int*)  carve((size_t)BN * KNN * 4);
  float* ef2  = (float*)carve((size_t)BN * KNN * 3 * 4);
  float* bufA = (float*)carve((size_t)BN * FDIM * 4);
  float* bufB = (float*)carve((size_t)BN * FDIM * 4);
  _Float16* f1h = (_Float16*)carve((size_t)BN * FDIM * 2);
  _Float16* f2h = (_Float16*)carve((size_t)BN * FDIM * 2);
  float* avec = (float*)carve((size_t)BN * 4);
  float* bvec = (float*)carve((size_t)BN * 4);
  float* kta  = (float*)carve((size_t)BN * 4);
  float* otfl = (float*)carve((size_t)BN * 3 * 4);

  // transposed/padded f16 weights: {d_in W index, K, C, Kpad}
  const int wdesc[18][4] = {
      {4, 6, 32, 32},   {6, 32, 32, 32},   {8, 32, 32, 32},
      {10, 35, 64, 64}, {12, 64, 64, 64},  {14, 64, 64, 64},
      {16, 67, 128, 96},{18, 128, 128, 128},{20, 128, 128, 128},
      {22, 6, 32, 32},  {24, 32, 32, 32},  {26, 32, 32, 32},
      {28, 35, 64, 64}, {30, 64, 64, 64},  {32, 64, 64, 64},
      {34, 67, 128, 96},{36, 128, 128, 128},{38, 128, 128, 128}};
  _Float16* wt[18];
  for (int w = 0; w < 18; ++w)
    wt[w] = (_Float16*)carve((size_t)wdesc[w][2] * wdesc[w][3] * 2);

  float* Kmat = (float*)carve((size_t)BN * NPTS * 4);   // 134 MB
  if (ws_size < off) return;  // workspace too small: bail safely

  // ---- weight prep ----
  for (int w = 0; w < 18; ++w) {
    int K = wdesc[w][1], C = wdesc[w][2], Kp = wdesc[w][3];
    int total = C * Kp;
    transw_kernel<<<(total + 255) / 256, 256, 0, stream>>>(
        (const float*)d_in[wdesc[w][0]], wt[w], K, C, Kp);
  }

  // ---- KNN graphs ----
  knn_kernel<<<BN / 128, 128, 0, stream>>>(pc1, idx1, ef1);
  knn_kernel<<<BN / 128, 128, 0, stream>>>(pc2, idx2, ef2);

  // ---- setconv stack helper ----
  auto setconv = [&](const float* x, int Cin, const int* idx, const float* ef,
                     int wbase, int Cout, float* tmp, float* pong) -> float* {
    const float* b1 = (const float*)d_in[wdesc[wbase][0] + 1];
    const float* b2 = (const float*)d_in[wdesc[wbase + 1][0] + 1];
    const float* b3 = (const float*)d_in[wdesc[wbase + 2][0] + 1];
    fc1max_kernel<<<BN / 8, 256, 0, stream>>>(x, Cin, idx, ef, wt[wbase],
                                              wdesc[wbase][3], b1, Cout, tmp);
    gemm_lrelu_kernel<<<BN / 16 / 8, 256, 0, stream>>>(tmp, Cout, wt[wbase + 1],
                                                       b2, Cout, pong);
    gemm_lrelu_kernel<<<BN / 16 / 8, 256, 0, stream>>>(pong, Cout, wt[wbase + 2],
                                                       b3, Cout, tmp);
    return tmp;
  };

  // ---- features for pc1 -> f1h ----
  float* h;
  h = setconv(pc1, 3, idx1, ef1, 0, 32, bufA, bufB);    // -> bufA [BN,32]
  h = setconv(h, 32, idx1, ef1, 3, 64, bufB, bufA);     // -> bufB [BN,64]
  h = setconv(h, 64, idx1, ef1, 6, 128, bufA, bufB);    // -> bufA [BN,128]
  normalize_kernel<<<BN / 4, 128, 0, stream>>>(h, f1h);

  // ---- features for pc2 -> f2h ----
  h = setconv(pc2, 3, idx2, ef2, 0, 32, bufA, bufB);
  h = setconv(h, 32, idx2, ef2, 3, 64, bufB, bufA);
  h = setconv(h, 64, idx2, ef2, 6, 128, bufA, bufB);
  normalize_kernel<<<BN / 4, 128, 0, stream>>>(h, f2h);

  // ---- transport kernel matrix (register-blocked WMMA) ----
  cost_kernel<<<(BATCH * 8192) / 8, 256, 0, stream>>>(f1h, f2h, pc1, pc2,
                                                      p_eps, Kmat);

  // ---- Sinkhorn (3 iterations) ----
  set_const_kernel<<<BN / 256, 256, 0, stream>>>(avec, 1.0f / NPTS, BN);
  for (int it = 0; it < 3; ++it) {
    hipMemsetAsync(kta, 0, (size_t)BN * 4, stream);
    colsum_kernel<<<BATCH * 32 * 4, 256, 0, stream>>>(Kmat, avec, kta);
    bupdate_kernel<<<BN / 256, 256, 0, stream>>>(kta, bvec, p_eps, p_gam);
    rowa_kernel<<<BN, 256, 0, stream>>>(Kmat, bvec, avec, p_eps, p_gam);
  }

  // ---- OT flow ----
  flow_kernel<<<BN, 256, 0, stream>>>(Kmat, avec, bvec, pc1, pc2, otfl);

  // ---- refinement convs on ot_flow (uses pc1 graph) ----
  h = setconv(otfl, 3, idx1, ef1, 9, 32, bufA, bufB);
  h = setconv(h, 32, idx1, ef1, 12, 64, bufB, bufA);
  h = setconv(h, 64, idx1, ef1, 15, 128, bufA, bufB);

  // ---- final fc + residual ----
  finalfc_kernel<<<BN / 256, 256, 0, stream>>>(h, (const float*)d_in[40],
                                               (const float*)d_in[41], otfl,
                                               (float*)d_out);
}
